// DecoderRNN_24842090840273
// MI455X (gfx1250) — matrix operand order repaired and checked by reading, not compiled, not run on hardware
//
#include <hip/hip_runtime.h>
#include <math.h>

// ---------- CDNA5 WMMA types ----------
typedef __attribute__((ext_vector_type(8)))  float  v8f;
typedef __attribute__((ext_vector_type(16))) __bf16 v16bf;

union BFrag {
    v16bf          v;
    unsigned short u[16];
    uint4          q[2];
};

__device__ __forceinline__ unsigned short f2bf(float f) {
    union { float f; unsigned u; } x; x.f = f;
    unsigned r = x.u + 0x7FFFu + ((x.u >> 16) & 1u);   // round-to-nearest-even
    return (unsigned short)(r >> 16);
}
__device__ __forceinline__ float sigm(float x) { return 1.0f / (1.0f + expf(-x)); }

// ---------------------------------------------------------------------------
// Generic bf16 WMMA GEMM:  C[M,N] = A[M,K] * W[N,K]^T + bias
//   A row-major bf16 (lda), W row-major bf16 (ldw) -> W is B^T, so each lane's
//   16 B elements are two 16-byte loads of 8 K-contiguous values each.
//   One wave computes a 32x64 tile: 2 A-fragments x 4 B-fragments = 8 WMMA
//   per K-step, each B load shared by both A tiles.
//   Out-of-range columns are CLAMPED for loads (branchless, EXEC stays all-1s
//   for WMMA) and guarded only at the store.
// ---------------------------------------------------------------------------
__global__ void __launch_bounds__(256)
gemm_bf16_wmma(const unsigned short* __restrict__ A, int lda,
               const unsigned short* __restrict__ W, int ldw,
               const float* __restrict__ bias,
               float* __restrict__ C, long long ldc,
               int M, int N, int K)
{
    const int m_pairs  = M >> 5;                 // 32 rows per wave
    const int n_groups = (N + 63) >> 6;          // 64 cols per wave
    const int gw = blockIdx.x * (blockDim.x >> 5) + (threadIdx.x >> 5);
    if (gw >= m_pairs * n_groups) return;        // uniform per wave
    const int mp = gw % m_pairs;
    const int ng = gw / m_pairs;
    const int m0 = mp << 5;
    const int n0 = ng << 6;

    const int lane = threadIdx.x & 31;
    const int lrow = lane & 15;
    const int hi   = lane >> 4;       // lane half select
    const int khA  = hi << 3;         // A halves: K 0..7 / 8..15 (and +16)
    const int khB  = hi << 3;         // B halves: K 0..7 / 8..15 (and +16)

    v8f acc[2][4];
    #pragma unroll
    for (int m = 0; m < 2; ++m)
        #pragma unroll
        for (int j = 0; j < 4; ++j)
            #pragma unroll
            for (int i = 0; i < 8; ++i) acc[m][j][i] = 0.0f;

    const unsigned short* ap0 = A + (long long)(m0 + lrow) * lda + khA;
    const unsigned short* ap1 = ap0 + (long long)16 * lda;

    // Hoisted, clamped B row pointers (K-invariant).
    const unsigned short* wp[4];
    #pragma unroll
    for (int j = 0; j < 4; ++j) {
        int col  = n0 + (j << 4) + lrow;
        int colL = (col < N) ? col : (N - 1);
        wp[j] = W + (long long)colL * ldw + khB;
    }

    for (int k0 = 0; k0 < K; k0 += 32) {
        BFrag a0, a1;
        a0.q[0] = *(const uint4*)(ap0 + k0);
        a0.q[1] = *(const uint4*)(ap0 + k0 + 16);
        a1.q[0] = *(const uint4*)(ap1 + k0);
        a1.q[1] = *(const uint4*)(ap1 + k0 + 16);

        #pragma unroll
        for (int j = 0; j < 4; ++j) {
            BFrag b;
            b.q[0] = *(const uint4*)(wp[j] + k0);
            b.q[1] = *(const uint4*)(wp[j] + k0 + 16);
            acc[0][j] = __builtin_amdgcn_wmma_f32_16x16x32_bf16(
                false, a0.v, false, b.v, (short)0, acc[0][j], false, false);
            acc[1][j] = __builtin_amdgcn_wmma_f32_16x16x32_bf16(
                false, a1.v, false, b.v, (short)0, acc[1][j], false, false);
        }
    }

    // C/D layout: VGPR i -> row m0 + i (+8 for upper lane half), col = lane&15
    const int rowoff = hi << 3;
    #pragma unroll
    for (int m = 0; m < 2; ++m) {
        const int mbase = m0 + (m << 4) + rowoff;
        #pragma unroll
        for (int j = 0; j < 4; ++j) {
            const int col = n0 + (j << 4) + lrow;
            if (col < N) {
                const float bv = bias ? bias[col] : 0.0f;
                #pragma unroll
                for (int i = 0; i < 8; ++i)
                    C[(long long)(mbase + i) * ldc + col] = acc[m][j][i] + bv;
            }
        }
    }
}

// ---------------------------------------------------------------------------
// Attention + lstm-input assembly (one block per batch row b)
//   scores[n] = A_w . tanh(u_hs[b,n,:] + w_ah[b,:]) + A_b ; softmax over N=49
//   context  = sum_n alpha[n] * features[b,n,:]
//   lstm_in[b] = bf16([ x_t | context | h_{t-1} | 0-pad ])   (K = 2880)
// ---------------------------------------------------------------------------
__global__ void __launch_bounds__(256)
attention_step(const float* __restrict__ u_hs,          // [B*49, 512]
               const float* __restrict__ w_ah,          // [B, 512]
               const float* __restrict__ A_w,           // [512]
               const float* __restrict__ A_b,           // [1]
               const float* __restrict__ features,      // [B, 49, 2048]
               const unsigned short* __restrict__ x_bf, // [B, 300] (this t)
               const unsigned short* __restrict__ h_bf, // [B, 512]
               unsigned short* __restrict__ lstm_in,    // [B, 2880]
               float* __restrict__ alphas,              // base + t*49; [b*ld + n]
               int ldalpha)
{
    __shared__ float s_wah[512];
    __shared__ float s_sc[49];
    __shared__ float s_alpha[49];
    const int b   = blockIdx.x;
    const int tid = threadIdx.x;

    for (int e = tid; e < 512; e += 256) s_wah[e] = w_ah[b * 512 + e];
    __syncthreads();

    const int wv = tid >> 5, lane = tid & 31;
    for (int n = wv; n < 49; n += 8) {
        const float* up = u_hs + (long long)(b * 49 + n) * 512;
        float p = 0.0f;
        for (int e = lane; e < 512; e += 32)
            p += A_w[e] * tanhf(up[e] + s_wah[e]);
        #pragma unroll
        for (int off = 16; off > 0; off >>= 1) p += __shfl_xor(p, off, 32);
        if (lane == 0) s_sc[n] = p + A_b[0];
    }
    __syncthreads();

    if (wv == 0) {  // softmax over 49 in one wave
        float v0 = (lane < 49)      ? s_sc[lane]      : -3.4e38f;
        float v1 = (lane + 32 < 49) ? s_sc[lane + 32] : -3.4e38f;
        float mx = fmaxf(v0, v1);
        #pragma unroll
        for (int off = 16; off > 0; off >>= 1) mx = fmaxf(mx, __shfl_xor(mx, off, 32));
        float e0 = (lane < 49)      ? expf(v0 - mx) : 0.0f;
        float e1 = (lane + 32 < 49) ? expf(v1 - mx) : 0.0f;
        float s  = e0 + e1;
        #pragma unroll
        for (int off = 16; off > 0; off >>= 1) s += __shfl_xor(s, off, 32);
        const float inv = 1.0f / s;
        if (lane < 49)      { s_alpha[lane]      = e0 * inv; alphas[b * ldalpha + lane]      = e0 * inv; }
        if (lane + 32 < 49) { s_alpha[lane + 32] = e1 * inv; alphas[b * ldalpha + lane + 32] = e1 * inv; }
    }
    __syncthreads();

    unsigned short* li = lstm_in + (long long)b * 2880;
    const float* fb = features + (long long)b * 49 * 2048;
    for (int e = tid; e < 2048; e += 256) {
        float acc = 0.0f;
        #pragma unroll 7
        for (int n = 0; n < 49; ++n) acc += s_alpha[n] * fb[n * 2048 + e];
        li[300 + e] = f2bf(acc);
    }
    for (int e = tid; e < 300; e += 256) li[e]        = x_bf[b * 300 + e];
    for (int e = tid; e < 512; e += 256) li[2348 + e] = h_bf[b * 512 + e];
    if (tid < 20) li[2860 + tid] = 0;   // zero K-pad
}

// ---------------------------------------------------------------------------
// LSTM pointwise: gates[b, 0:512|512:1024|1024:1536|1536:2048] = i|f|g|o
// ---------------------------------------------------------------------------
__global__ void __launch_bounds__(256)
lstm_pointwise(const float* __restrict__ gates,
               float* __restrict__ h, float* __restrict__ c,
               unsigned short* __restrict__ h_bf)
{
    const int idx = blockIdx.x * blockDim.x + threadIdx.x;
    if (idx >= 64 * 512) return;
    const int b = idx >> 9, d = idx & 511;
    const float* g = gates + b * 2048;
    const float ig = sigm(g[d]);
    const float fg = sigm(g[512 + d]);
    const float gg = tanhf(g[1024 + d]);
    const float og = sigm(g[1536 + d]);
    const float cn = fg * c[idx] + ig * gg;
    const float hn = og * tanhf(cn);
    c[idx] = cn; h[idx] = hn; h_bf[idx] = f2bf(hn);
}

// ------------------------- one-time prep kernels ---------------------------
__global__ void convert_f32_bf16(const float* __restrict__ s,
                                 unsigned short* __restrict__ d, long long n)
{
    long long i = (long long)blockIdx.x * blockDim.x + threadIdx.x;
    const long long stride = (long long)gridDim.x * blockDim.x;
    for (; i < n; i += stride) d[i] = f2bf(s[i]);
}

// Wg[2048, 2880] = bf16([ ih_w(2348) | hh_w(512) | 0-pad(20) ]) per row
__global__ void build_gate_weight(const float* __restrict__ ih_w,
                                  const float* __restrict__ hh_w,
                                  unsigned short* __restrict__ Wg)
{
    long long i = (long long)blockIdx.x * blockDim.x + threadIdx.x;
    const long long total  = 2048LL * 2880;
    const long long stride = (long long)gridDim.x * blockDim.x;
    for (; i < total; i += stride) {
        const int r = (int)(i / 2880), cc = (int)(i % 2880);
        float v;
        if (cc < 2348)      v = ih_w[(long long)r * 2348 + cc];
        else if (cc < 2860) v = hh_w[(long long)r * 512 + (cc - 2348)];
        else                v = 0.0f;
        Wg[i] = f2bf(v);
    }
}

__global__ void build_gate_bias(const float* __restrict__ ih_b,
                                const float* __restrict__ hh_b,
                                float* __restrict__ gb)
{
    const int i = blockIdx.x * blockDim.x + threadIdx.x;
    if (i < 2048) gb[i] = ih_b[i] + hh_b[i];
}

// x_bf layout [T=30][B=64][300]
__global__ void gather_embeds(const int* __restrict__ captions,   // [B, 31]
                              const float* __restrict__ emb,      // [V, 300]
                              unsigned short* __restrict__ x_bf)
{
    long long i = (long long)blockIdx.x * blockDim.x + threadIdx.x;
    const long long total  = 30LL * 64 * 300;
    const long long stride = (long long)gridDim.x * blockDim.x;
    for (; i < total; i += stride) {
        const int e  = (int)(i % 300);
        const int rb = (int)(i / 300);
        const int b  = rb & 63;
        const int t  = rb >> 6;
        const int tok = captions[b * 31 + t];
        x_bf[i] = f2bf(emb[(long long)tok * 300 + e]);
    }
}

__global__ void mean_features(const float* __restrict__ f,
                              unsigned short* __restrict__ mean_bf)
{
    const int idx = blockIdx.x * blockDim.x + threadIdx.x;  // 64*2048
    if (idx >= 64 * 2048) return;
    const int b = idx >> 11, e = idx & 2047;
    const float* p = f + (long long)b * 49 * 2048 + e;
    float s = 0.0f;
    for (int n = 0; n < 49; ++n) s += p[n * 2048];
    mean_bf[idx] = f2bf(s * (1.0f / 49.0f));
}

__global__ void hbf_from_h(const float* __restrict__ h,
                           unsigned short* __restrict__ h_bf)
{
    const int idx = blockIdx.x * blockDim.x + threadIdx.x;
    if (idx < 64 * 512) h_bf[idx] = f2bf(h[idx]);
}

// ---------------------------------------------------------------------------
extern "C" void kernel_launch(void* const* d_in, const int* in_sizes, int n_in,
                              void* d_out, int out_size, void* d_ws, size_t ws_size,
                              hipStream_t stream)
{
    (void)in_sizes; (void)n_in; (void)out_size; (void)ws_size;

    const float* features = (const float*)d_in[0];   // [64,49,2048]
    const int*   captions = (const int*)d_in[1];     // [64,31]
    const float* emb      = (const float*)d_in[2];   // [10000,300]
    const float* U_w      = (const float*)d_in[3];   // [512,2048]
    const float* U_b      = (const float*)d_in[4];
    const float* W_w      = (const float*)d_in[5];   // [512,512]
    const float* W_b      = (const float*)d_in[6];
    const float* A_w      = (const float*)d_in[7];   // [1,512]
    const float* A_b      = (const float*)d_in[8];
    const float* initH_w  = (const float*)d_in[9];   // [512,2048]
    const float* initH_b  = (const float*)d_in[10];
    const float* initC_w  = (const float*)d_in[11];
    const float* initC_b  = (const float*)d_in[12];
    const float* ih_w     = (const float*)d_in[13];  // [2048,2348]
    const float* ih_b     = (const float*)d_in[14];
    const float* hh_w     = (const float*)d_in[15];  // [2048,512]
    const float* hh_b     = (const float*)d_in[16];
    const float* fcn_w    = (const float*)d_in[17];  // [10000,512]
    const float* fcn_b    = (const float*)d_in[18];

    float* preds  = (float*)d_out;                   // [64,30,10000]
    float* alphas = preds + 64LL * 30 * 10000;       // [64,30,49]

    // ---- carve workspace ----
    char* ws = (char*)d_ws;
    auto alloc = [&](size_t bytes) -> char* {
        char* p = ws; ws += (bytes + 255) & ~(size_t)255; return p;
    };
    unsigned short* features_bf = (unsigned short*)alloc(64LL * 49 * 2048 * 2);
    unsigned short* U_bf        = (unsigned short*)alloc(512LL * 2048 * 2);
    unsigned short* W_bf        = (unsigned short*)alloc(512LL * 512 * 2);
    unsigned short* initH_bf    = (unsigned short*)alloc(512LL * 2048 * 2);
    unsigned short* initC_bf    = (unsigned short*)alloc(512LL * 2048 * 2);
    unsigned short* Wg_bf       = (unsigned short*)alloc(2048LL * 2880 * 2);
    unsigned short* fcn_bf      = (unsigned short*)alloc(10000LL * 512 * 2);
    unsigned short* mean_bf     = (unsigned short*)alloc(64LL * 2048 * 2);
    unsigned short* x_bf        = (unsigned short*)alloc(30LL * 64 * 300 * 2);
    float*          u_hs        = (float*)alloc(3136LL * 512 * 4);
    float*          w_ah        = (float*)alloc(64LL * 512 * 4);
    float*          hbuf        = (float*)alloc(64LL * 512 * 4);
    float*          cbuf        = (float*)alloc(64LL * 512 * 4);
    unsigned short* h_bf        = (unsigned short*)alloc(64LL * 512 * 2);
    float*          gbias       = (float*)alloc(2048LL * 4);
    unsigned short* lstm_in     = (unsigned short*)alloc(64LL * 2880 * 2);
    float*          gates       = (float*)alloc(64LL * 2048 * 4);

    auto gemm = [&](const unsigned short* A, int lda,
                    const unsigned short* Wm, int ldw, const float* bias,
                    float* C, long long ldc, int M, int N, int K) {
        const int waves  = (M >> 5) * ((N + 63) >> 6);
        const int blocks = (waves + 7) / 8;
        gemm_bf16_wmma<<<blocks, 256, 0, stream>>>(A, lda, Wm, ldw, bias, C, ldc, M, N, K);
    };

    // ---- one-time prep (runs inside the captured graph every replay) ----
    convert_f32_bf16<<<1024, 256, 0, stream>>>(features, features_bf, 64LL * 49 * 2048);
    convert_f32_bf16<<<512,  256, 0, stream>>>(U_w,     U_bf,     512LL * 2048);
    convert_f32_bf16<<<256,  256, 0, stream>>>(W_w,     W_bf,     512LL * 512);
    convert_f32_bf16<<<512,  256, 0, stream>>>(initH_w, initH_bf, 512LL * 2048);
    convert_f32_bf16<<<512,  256, 0, stream>>>(initC_w, initC_bf, 512LL * 2048);
    convert_f32_bf16<<<1024, 256, 0, stream>>>(fcn_w,   fcn_bf,   10000LL * 512);
    build_gate_weight<<<1024, 256, 0, stream>>>(ih_w, hh_w, Wg_bf);
    build_gate_bias<<<8, 256, 0, stream>>>(ih_b, hh_b, gbias);
    gather_embeds<<<1024, 256, 0, stream>>>(captions, emb, x_bf);
    mean_features<<<512, 256, 0, stream>>>(features, mean_bf);

    // u_hs = features @ U_w^T + U_b   [3136, 512]
    gemm(features_bf, 2048, U_bf, 2048, U_b, u_hs, 512, 3136, 512, 2048);
    // h0 / c0
    gemm(mean_bf, 2048, initH_bf, 2048, initH_b, hbuf, 512, 64, 512, 2048);
    gemm(mean_bf, 2048, initC_bf, 2048, initC_b, cbuf, 512, 64, 512, 2048);
    hbf_from_h<<<128, 256, 0, stream>>>(hbuf, h_bf);

    // ---- 30 decode steps ----
    for (int t = 0; t < 30; ++t) {
        // w_ah = h @ W_w^T + W_b   [64, 512]
        gemm(h_bf, 512, W_bf, 512, W_b, w_ah, 512, 64, 512, 512);
        // attention + assemble lstm_in
        attention_step<<<64, 256, 0, stream>>>(u_hs, w_ah, A_w, A_b, features,
                                               x_bf + (long long)t * 64 * 300,
                                               h_bf, lstm_in,
                                               alphas + (long long)t * 49, 30 * 49);
        // gates = [x|ctx|h] @ [ih_w|hh_w]^T + (ih_b + hh_b)   [64, 2048]
        gemm(lstm_in, 2880, Wg_bf, 2880, gbias, gates, 2048, 64, 2048, 2880);
        // LSTM cell pointwise
        lstm_pointwise<<<128, 256, 0, stream>>>(gates, hbuf, cbuf, h_bf);
        // preds[:, t, :] = h @ fcn_w^T + fcn_b   [64, 10000]
        gemm(h_bf, 512, fcn_bf, 512, fcn_b,
             preds + (long long)t * 10000, 30LL * 10000, 64, 10000, 512);
    }
}
